// GroupedQueryAttention_44805098831958
// MI455X (gfx1250) — matrix operand order repaired and checked by reading, not compiled, not run on hardware
//
#include <hip/hip_runtime.h>

typedef unsigned short u16;
typedef unsigned int   u32;
typedef __attribute__((ext_vector_type(16))) __bf16 v16bf;
typedef __attribute__((ext_vector_type(8)))  float  v8f;

#define B_    2
#define T_    2048
#define C_    2048
#define H_    16
#define KVH_  4
#define D_    128
#define SCALE_ 0.08838834764831845f   // 1/sqrt(128)

union Frag {
  u32   u[8];
  v16bf v;
};

__device__ __forceinline__ u16 f2bf(float f) {
  u32 u = __float_as_uint(f);
  u += 0x7FFFu + ((u >> 16) & 1u);   // round-to-nearest-even
  return (u16)(u >> 16);
}
__device__ __forceinline__ float bf2f(u16 h) {
  return __uint_as_float(((u32)h) << 16);
}
__device__ __forceinline__ v8f v8f_zero() {
  v8f z = {0.f, 0.f, 0.f, 0.f, 0.f, 0.f, 0.f, 0.f};
  return z;
}
__device__ __forceinline__ v8f wmma_bf16(const Frag& a, const Frag& b, v8f c) {
  return __builtin_amdgcn_wmma_f32_16x16x32_bf16(
      false, a.v, false, b.v, (short)0, c, false, false);
}

// ---------------------------------------------------------------- f32 -> bf16 (x4 vectorized)
__global__ __launch_bounds__(256) void cvt_kernel(const float* __restrict__ in,
                                                  u16* __restrict__ out, size_t n) {
  size_t base = ((size_t)blockIdx.x * 256 + threadIdx.x) * 4;
  if (base < n) {  // n is always a multiple of 4 here
    float4 f = *(const float4*)(in + base);
    uint2 r;
    r.x = (u32)f2bf(f.x) | ((u32)f2bf(f.y) << 16);
    r.y = (u32)f2bf(f.z) | ((u32)f2bf(f.w) << 16);
    *(uint2*)(out + base) = r;
  }
}

// ---------------------------------------------------------------- bf16 GEMM
// C[M,N] = A[M,K] * B[K,N], row-major, v_wmma_f32_16x16x32_bf16.
// 256 threads (8 waves), 128x128 tile, K-step 32, double-buffered LDS with
// register staging: next tile's global loads overlap current tile's WMMAs.
template <bool OUT_F32>
__global__ __launch_bounds__(256) void gemm_bf16_kernel(
    const u16* __restrict__ A, const u16* __restrict__ Bm,
    void* __restrict__ Cout, int M, int N, int K) {
  __shared__ u32 AsP[2][128][17];  // AsP[b][m][c] = {A[m][2c], A[m][2c+1]}
  __shared__ u32 BsP[2][128][17];  // BsP[b][n][kk] = {B[2kk][n], B[2kk+1][n]}

  const int tid  = threadIdx.x;
  const int lane = tid & 31;
  const int wave = tid >> 5;
  const int wm   = wave >> 1;   // 0..3
  const int wn   = wave & 1;    // 0..1
  const int lr   = lane & 15;
  const int half = lane >> 4;
  const int m0 = blockIdx.y * 128;
  const int n0 = blockIdx.x * 128;

  u32 ra[8];        // staged A tile (8 dwords/thread)
  u32 rb[4][2];     // staged B tile (4 x 2 dwords/thread)

  auto load_tiles = [&](int k0) {
#pragma unroll
    for (int t = 0; t < 8; ++t) {
      int i = tid + t * 256;
      int r = i >> 4, c = i & 15;
      ra[t] = *(const u32*)(A + (size_t)(m0 + r) * K + (k0 + c * 2));
    }
#pragma unroll
    for (int t = 0; t < 4; ++t) {
      int i = tid + t * 256;
      int kk = i >> 6, nn = i & 63;
      const u16* p0 = Bm + (size_t)(k0 + 2 * kk) * N + (n0 + 2 * nn);
      rb[t][0] = *(const u32*)(p0);
      rb[t][1] = *(const u32*)(p0 + N);
    }
  };
  auto store_tiles = [&](int buf) {
#pragma unroll
    for (int t = 0; t < 8; ++t) {
      int i = tid + t * 256;
      AsP[buf][i >> 4][i & 15] = ra[t];
    }
#pragma unroll
    for (int t = 0; t < 4; ++t) {
      int i = tid + t * 256;
      int kk = i >> 6, nn = i & 63;
      u32 a0 = rb[t][0], a1 = rb[t][1];
      BsP[buf][2 * nn][kk]     = (a0 & 0xFFFFu) | (a1 << 16);
      BsP[buf][2 * nn + 1][kk] = (a0 >> 16)     | (a1 & 0xFFFF0000u);
    }
  };

  v8f acc[2][4];
#pragma unroll
  for (int i = 0; i < 2; ++i)
#pragma unroll
    for (int j = 0; j < 4; ++j) acc[i][j] = v8f_zero();

  load_tiles(0);
  store_tiles(0);
  __syncthreads();

  const int nk = K / 32;
  for (int kt = 0; kt < nk; ++kt) {
    const int cur = kt & 1;
    const bool has_next = (kt + 1) < nk;
    if (has_next) load_tiles((kt + 1) * 32);            // overlap with WMMAs below
    if (kt + 2 < nk) {                                  // pull tile kt+2 toward L2
      int i = tid;
      __builtin_prefetch(A + (size_t)(m0 + (i >> 4)) * K + ((kt + 2) * 32 + (i & 15) * 2), 0, 1);
      __builtin_prefetch(Bm + (size_t)((kt + 2) * 32 + 2 * (i >> 6)) * N + (n0 + 2 * (i & 63)), 0, 1);
    }

    Frag bf[4];
#pragma unroll
    for (int nt = 0; nt < 4; ++nt) {
      int n = wn * 64 + nt * 16 + lr;
#pragma unroll
      for (int j = 0; j < 8; ++j) bf[nt].u[j] = BsP[cur][n][half * 8 + j];
    }
#pragma unroll
    for (int mt = 0; mt < 2; ++mt) {
      Frag af;
      int r = wm * 32 + mt * 16 + lr;
#pragma unroll
      for (int j = 0; j < 8; ++j)
        af.u[j] = AsP[cur][r][half * 8 + j];
#pragma unroll
      for (int nt = 0; nt < 4; ++nt) acc[mt][nt] = wmma_bf16(af, bf[nt], acc[mt][nt]);
    }

    if (has_next) store_tiles(cur ^ 1);
    __syncthreads();
  }

#pragma unroll
  for (int mt = 0; mt < 2; ++mt)
#pragma unroll
    for (int nt = 0; nt < 4; ++nt)
#pragma unroll
      for (int r = 0; r < 8; ++r) {
        int row = m0 + wm * 32 + mt * 16 + r + half * 8;
        int col = n0 + wn * 64 + nt * 16 + lr;
        float v = acc[mt][nt][r];
        if (OUT_F32) ((float*)Cout)[(size_t)row * N + col] = v;
        else         ((u16*)Cout)[(size_t)row * N + col] = f2bf(v);
      }
}

// ---------------------------------------------------------------- RoPE (in-place, bf16)
// x layout: (B*T, H*D); pair i = dims (2i, 2i+1) -> one b32 per pair; cos/sin: (T, 64)
__global__ __launch_bounds__(256) void rope_kernel(u16* __restrict__ x,
                                                   const float* __restrict__ cosb,
                                                   const float* __restrict__ sinb,
                                                   int H, int total_pairs) {
  int idx = blockIdx.x * 256 + threadIdx.x;
  if (idx >= total_pairs) return;
  int d2 = idx & 63;
  int t  = (idx / (64 * H)) % T_;
  float c = cosb[t * 64 + d2];
  float s = sinb[t * 64 + d2];
  u32* p = (u32*)x + idx;
  u32 v = *p;
  float x1 = bf2f((u16)(v & 0xFFFFu));
  float x2 = bf2f((u16)(v >> 16));
  u16 y1 = f2bf(x1 * c - x2 * s);
  u16 y2 = f2bf(x1 * s + x2 * c);
  *p = (u32)y1 | ((u32)y2 << 16);
}

// ---------------------------------------------------------------- flash attention
// grid: (T/128, B*H). Each wave: 16 q-rows, full D=128 in registers.
// K/V blocks double-buffered in LDS with register staging.
__global__ __launch_bounds__(256) void attn_kernel(const u16* __restrict__ qb,
                                                   const u16* __restrict__ kb,
                                                   const u16* __restrict__ vb,
                                                   u16* __restrict__ ob) {
  __shared__ u32 Ks[2][32][68];    // 32 keys x 128 bf16 (64 dwords) + pad
  __shared__ u32 VsP[2][128][17];  // VsP[b][d][kk] = {V[2kk][d], V[2kk+1][d]}
  __shared__ u16 Ps[8][16][34];    // per-wave P scratch (C-layout -> A-layout)

  const int tid  = threadIdx.x;
  const int lane = tid & 31;
  const int w    = tid >> 5;
  const int lr   = lane & 15;
  const int half = lane >> 4;

  const int qrow0 = blockIdx.x * 128;
  const int bidx  = blockIdx.y >> 4;
  const int h     = blockIdx.y & 15;
  const int kvh   = h >> 2;

  const size_t kvrow0 = (size_t)(bidx * T_) * (KVH_ * D_) + (size_t)kvh * D_;

  uint4 rk[2];      // staged K block
  u32   rv[4][2];   // staged V block

  auto load_kv = [&](int kb0) {
#pragma unroll
    for (int t = 0; t < 2; ++t) {
      int i = tid + t * 256;
      int key = i >> 4, c4 = i & 15;
      rk[t] = *(const uint4*)(kb + kvrow0 + (size_t)(kb0 + key) * (KVH_ * D_) + c4 * 8);
    }
#pragma unroll
    for (int t = 0; t < 4; ++t) {
      int i = tid + t * 256;
      int kk = i >> 6, nn = i & 63;
      const u16* p0 = vb + kvrow0 + (size_t)(kb0 + 2 * kk) * (KVH_ * D_) + 2 * nn;
      rv[t][0] = *(const u32*)(p0);
      rv[t][1] = *(const u32*)(p0 + KVH_ * D_);
    }
  };
  auto store_kv = [&](int buf) {
#pragma unroll
    for (int t = 0; t < 2; ++t) {
      int i = tid + t * 256;
      *(uint4*)&Ks[buf][i >> 4][(i & 15) * 4] = rk[t];
    }
#pragma unroll
    for (int t = 0; t < 4; ++t) {
      int i = tid + t * 256;
      int kk = i >> 6, nn = i & 63;
      u32 a0 = rv[t][0], a1 = rv[t][1];
      VsP[buf][2 * nn][kk]     = (a0 & 0xFFFFu) | (a1 << 16);
      VsP[buf][2 * nn + 1][kk] = (a0 >> 16)     | (a1 & 0xFFFF0000u);
    }
  };

  // Q fragments: row = qrow0 + w*16 + lr (A-layout m), 4 frags over D=128
  const size_t qbase =
      (size_t)(bidx * T_ + qrow0 + w * 16 + lr) * (H_ * D_) + (size_t)h * D_;
  Frag qa[4];
#pragma unroll
  for (int f = 0; f < 4; ++f)
#pragma unroll
    for (int g = 0; g < 2; ++g) {
      uint4 q4 = *(const uint4*)(qb + qbase + f * 32 + half * 16 + g * 8);
      qa[f].u[g * 4 + 0] = q4.x;
      qa[f].u[g * 4 + 1] = q4.y;
      qa[f].u[g * 4 + 2] = q4.z;
      qa[f].u[g * 4 + 3] = q4.w;
    }

  v8f o[8];
  float mrun[8], lrun[8];
#pragma unroll
  for (int i = 0; i < 8; ++i) {
    o[i] = v8f_zero();
    mrun[i] = -1e30f;
    lrun[i] = 0.f;
  }

  load_kv(0);
  store_kv(0);
  __syncthreads();

  const int nkb = (qrow0 + 128) / 32;
  for (int it = 0; it < nkb; ++it) {
    const int kb0 = it * 32;
    const int cur = it & 1;
    const bool has_next = (it + 1) < nkb;
    if (has_next) load_kv(kb0 + 32);                    // overlap with WMMAs below
    if (it + 2 < nkb) {                                 // pull block it+2 toward L2
      __builtin_prefetch(kb + kvrow0 + (size_t)(kb0 + 64 + (tid >> 4)) * (KVH_ * D_) + (tid & 15) * 8, 0, 1);
      __builtin_prefetch(vb + kvrow0 + (size_t)(kb0 + 64 + (tid >> 4)) * (KVH_ * D_) + (tid & 15) * 8, 0, 1);
    }

    // S = Q K^T : two 16-key tiles, accumulate over 4 d-chunks of 32
    v8f s0 = v8f_zero(), s1 = v8f_zero();
#pragma unroll
    for (int f = 0; f < 4; ++f) {
      Frag k0f, k1f;
#pragma unroll
      for (int j = 0; j < 8; ++j) {
        k0f.u[j] = Ks[cur][lr][f * 16 + half * 8 + j];
        k1f.u[j] = Ks[cur][lr + 16][f * 16 + half * 8 + j];
      }
      s0 = wmma_bf16(qa[f], k0f, s0);
      s1 = wmma_bf16(qa[f], k1f, s1);
    }

    // scale + causal mask + online softmax (row stats across 16-lane halves)
    float alpha[8];
#pragma unroll
    for (int r = 0; r < 8; ++r) {
      int row = qrow0 + w * 16 + r + half * 8;
      float v0 = s0[r] * SCALE_;
      float v1 = s1[r] * SCALE_;
      if (kb0 + lr > row)      v0 = -1e30f;
      if (kb0 + 16 + lr > row) v1 = -1e30f;
      float pm = fmaxf(v0, v1);
      pm = fmaxf(pm, __shfl_xor(pm, 1, 32));
      pm = fmaxf(pm, __shfl_xor(pm, 2, 32));
      pm = fmaxf(pm, __shfl_xor(pm, 4, 32));
      pm = fmaxf(pm, __shfl_xor(pm, 8, 32));
      float mnew = fmaxf(mrun[r], pm);
      alpha[r] = __expf(mrun[r] - mnew);
      mrun[r] = mnew;
      float p0 = __expf(v0 - mnew);
      float p1 = __expf(v1 - mnew);
      float ps = p0 + p1;
      ps += __shfl_xor(ps, 1, 32);
      ps += __shfl_xor(ps, 2, 32);
      ps += __shfl_xor(ps, 4, 32);
      ps += __shfl_xor(ps, 8, 32);
      lrun[r] = lrun[r] * alpha[r] + ps;
      Ps[w][r + half * 8][lr]      = f2bf(p0);
      Ps[w][r + half * 8][16 + lr] = f2bf(p1);
#pragma unroll
      for (int nt = 0; nt < 8; ++nt) o[nt][r] *= alpha[r];
    }

    // P as A-fragment (wave-private scratch; same-wave DS ops are in order)
    Frag pf;
#pragma unroll
    for (int j = 0; j < 8; ++j) {
      int d = half * 16 + j * 2;
      pf.u[j] = *(const u32*)&Ps[w][lr][d];
    }
    // O += P V
#pragma unroll
    for (int nt = 0; nt < 8; ++nt) {
      Frag vf;
#pragma unroll
      for (int j = 0; j < 8; ++j) vf.u[j] = VsP[cur][nt * 16 + lr][half * 8 + j];
      o[nt] = wmma_bf16(pf, vf, o[nt]);
    }

    if (has_next) store_kv(cur ^ 1);
    __syncthreads();
  }

  // normalize + store bf16 (B*T, H*D)
#pragma unroll
  for (int r = 0; r < 8; ++r) {
    int row = qrow0 + w * 16 + r + half * 8;
    float inv = 1.0f / lrun[r];
    size_t base = (size_t)(bidx * T_ + row) * (H_ * D_) + (size_t)h * D_ + lr;
#pragma unroll
    for (int nt = 0; nt < 8; ++nt) ob[base + nt * 16] = f2bf(o[nt][r] * inv);
  }
}

// ---------------------------------------------------------------- launcher
extern "C" void kernel_launch(void* const* d_in, const int* in_sizes, int n_in,
                              void* d_out, int out_size, void* d_ws, size_t ws_size,
                              hipStream_t stream) {
  (void)in_sizes; (void)n_in; (void)out_size; (void)ws_size;
  const float* x    = (const float*)d_in[0];
  const float* cosb = (const float*)d_in[1];
  const float* sinb = (const float*)d_in[2];
  // d_in[3] is the causal mask (computed analytically in-kernel)
  const float* Wq = (const float*)d_in[4];
  const float* Wk = (const float*)d_in[5];
  const float* Wv = (const float*)d_in[6];
  const float* Wo = (const float*)d_in[7];

  char* ws = (char*)d_ws;
  size_t off = 0;
  auto alloc = [&](size_t bytes) -> void* {
    void* p = ws + off;
    off += (bytes + 255) & ~(size_t)255;
    return p;
  };
  u16* xb   = (u16*)alloc((size_t)B_ * T_ * C_ * 2);
  u16* wqb  = (u16*)alloc((size_t)C_ * H_ * D_ * 2);
  u16* wkb  = (u16*)alloc((size_t)C_ * KVH_ * D_ * 2);
  u16* wvb  = (u16*)alloc((size_t)C_ * KVH_ * D_ * 2);
  u16* wob  = (u16*)alloc((size_t)H_ * D_ * C_ * 2);
  u16* qbuf = (u16*)alloc((size_t)B_ * T_ * H_ * D_ * 2);
  u16* kbuf = (u16*)alloc((size_t)B_ * T_ * KVH_ * D_ * 2);
  u16* vbuf = (u16*)alloc((size_t)B_ * T_ * KVH_ * D_ * 2);
  u16* obuf = (u16*)alloc((size_t)B_ * T_ * H_ * D_ * 2);

  auto cvt = [&](const float* in, u16* out, size_t n) {
    cvt_kernel<<<(unsigned)((n / 4 + 255) / 256), 256, 0, stream>>>(in, out, n);
  };
  cvt(x,  xb,  (size_t)B_ * T_ * C_);
  cvt(Wq, wqb, (size_t)C_ * H_ * D_);
  cvt(Wk, wkb, (size_t)C_ * KVH_ * D_);
  cvt(Wv, wvb, (size_t)C_ * KVH_ * D_);
  cvt(Wo, wob, (size_t)H_ * D_ * C_);

  const int M = B_ * T_;
  gemm_bf16_kernel<false><<<dim3((H_ * D_) / 128, M / 128), 256, 0, stream>>>(
      xb, wqb, qbuf, M, H_ * D_, C_);
  gemm_bf16_kernel<false><<<dim3((KVH_ * D_) / 128, M / 128), 256, 0, stream>>>(
      xb, wkb, kbuf, M, KVH_ * D_, C_);
  gemm_bf16_kernel<false><<<dim3((KVH_ * D_) / 128, M / 128), 256, 0, stream>>>(
      xb, wvb, vbuf, M, KVH_ * D_, C_);

  rope_kernel<<<(B_ * T_ * H_ * 64) / 256, 256, 0, stream>>>(
      qbuf, cosb, sinb, H_, B_ * T_ * H_ * 64);
  rope_kernel<<<(B_ * T_ * KVH_ * 64) / 256, 256, 0, stream>>>(
      kbuf, cosb, sinb, KVH_, B_ * T_ * KVH_ * 64);

  attn_kernel<<<dim3(T_ / 128, B_ * H_), 256, 0, stream>>>(qbuf, kbuf, vbuf, obuf);

  gemm_bf16_kernel<true><<<dim3(C_ / 128, M / 128), 256, 0, stream>>>(
      obuf, wob, d_out, M, C_, H_ * D_);
}
